// _SiameseGATv2Block_88089779241033
// MI455X (gfx1250) — matrix operand order repaired and checked by reading, not compiled, not run on hardware
//
#include <hip/hip_runtime.h>
#include <hip/hip_bf16.h>

// ---------------- problem constants ----------------
#define NN      100000
#define DD      128
#define HH      8
#define CC      16
#define E_RAW   1600000
#define EE      (E_RAW + NN)        // 1,700,000 edges incl. self loops
#define NTILES  (NN / 16)           // 6250 row tiles (exact)
#define NEG_SLOPE 0.2f
#define LN_EPS  1e-5f

typedef __attribute__((ext_vector_type(16))) __bf16          v16bf;
typedef __attribute__((ext_vector_type(16))) unsigned short  v16u;
typedef __attribute__((ext_vector_type(8)))  float           v8f;

// ---------------- helpers ----------------
__device__ __forceinline__ unsigned short f2bf(float f) {
    unsigned u = __float_as_uint(f);
    unsigned r = u + 0x7fffu + ((u >> 16) & 1u);   // RNE to bf16
    return (unsigned short)(r >> 16);
}
// order-preserving float -> uint for atomicMax
__device__ __forceinline__ unsigned encF(float f) {
    unsigned u = __float_as_uint(f);
    return (u & 0x80000000u) ? ~u : (u | 0x80000000u);
}
__device__ __forceinline__ float decF(unsigned u) {
    unsigned b = (u & 0x80000000u) ? (u ^ 0x80000000u) : ~u;
    return __uint_as_float(b);
}
#define ENC_NEG_INF 0x007FFFFFu   // encF(-inf)

__device__ __forceinline__ float lrelu(float v) { return v > 0.f ? v : NEG_SLOPE * v; }
__device__ __forceinline__ float elu1(float v)  { return v > 0.f ? v : (__expf(v) - 1.f); }

__device__ __forceinline__ void edge_sd(const long long* __restrict__ eidx, int e,
                                        int& s, int& d) {
    if (e < E_RAW) { s = (int)eidx[e]; d = (int)eidx[E_RAW + e]; }
    else           { s = d = e - E_RAW; }
}

// ---------------- kernel 1: pack 4 weight matrices (f32 row-major -> bf16 lane-major WMMA-B tiles) ----------------
// Per matrix: tile index (kt*8+nt), lane 0..31, 16 bf16 values.
// lane's 16 values = W[kt*32 + (v<8 ? (lane&16?8:0) + v : 16 + (lane&16?8:0) + v-8)][nt*16 + (lane&15)], v=0..15
__global__ void pack_w_kernel(const float* __restrict__ Wl0, const float* __restrict__ Wr0,
                              const float* __restrict__ Wl1, const float* __restrict__ Wr1,
                              unsigned short* __restrict__ wpack) {
    int t = blockIdx.x * 256 + threadIdx.x;            // 4*16384 total
    if (t >= 4 * 16384) return;
    int widx = t >> 14;
    int r    = t & 16383;
    int tile = r >> 9;                                  // 0..31
    int lane = (r >> 4) & 31;
    int v    = r & 15;
    int kt = tile >> 3, nt = tile & 7;
    int k = kt * 32 + ((lane & 16) ? 8 : 0) + (v < 8 ? v : v + 8);
    int n = nt * 16 + (lane & 15);
    const float* W = (widx == 0) ? Wl0 : (widx == 1) ? Wr0 : (widx == 2) ? Wl1 : Wr1;
    wpack[t] = f2bf(W[k * 128 + n]);
}

// ---------------- kernel 2: y = x @ W via bf16 WMMA, one wave per 16x128 output strip ----------------
__global__ void gemm_wmma_kernel(const float* __restrict__ x,
                                 const unsigned short* __restrict__ wpack,
                                 float* __restrict__ y) {
    int wave = blockIdx.x * (blockDim.x >> 5) + (threadIdx.x >> 5);
    if (wave >= NTILES) return;
    int lane = threadIdx.x & 31;
    int mrow = (wave << 4) + (lane & 15);
    int sel  = (lane & 16) ? 8 : 0;                     // K-half select per A layout
    const float* xrow = x + (size_t)mrow * DD;

    v8f acc[8];
#pragma unroll
    for (int i = 0; i < 8; ++i) acc[i] = (v8f){};

#pragma unroll
    for (int kt = 0; kt < 4; ++kt) {
        int kk = kt * 32;
        v8f alo = *(const v8f*)(xrow + kk + sel);        // K = kk+sel   .. +7
        v8f ahi = *(const v8f*)(xrow + kk + 16 + sel);   // K = kk+16+sel.. +7
        union { v16u u; v16bf b; } a;
#pragma unroll
        for (int i = 0; i < 8; ++i) { a.u[i] = f2bf(alo[i]); a.u[8 + i] = f2bf(ahi[i]); }
#pragma unroll
        for (int nt = 0; nt < 8; ++nt) {
            union { v16u u; v16bf b; } bm;
            bm.u = *(const v16u*)(wpack + ((kt * 8 + nt) * 512 + lane * 16));
            acc[nt] = __builtin_amdgcn_wmma_f32_16x16x32_bf16(
                false, a.b, false, bm.b, (short)0, acc[nt], false, false);
        }
    }
    // C/D layout: VGPR r -> (M = r + (lane&16?8:0), N = lane&15)
    int rbase = (wave << 4) + ((lane & 16) ? 8 : 0);
    int cbase = lane & 15;
#pragma unroll
    for (int nt = 0; nt < 8; ++nt)
#pragma unroll
        for (int r = 0; r < 8; ++r)
            y[(size_t)(rbase + r) * DD + nt * 16 + cbase] = acc[nt][r];
}

// ---------------- kernel 3: zero/init per-layer accumulators ----------------
__global__ void init_kernel(float* __restrict__ agg, float* __restrict__ denom,
                            unsigned* __restrict__ emaxU) {
    int t = blockIdx.x * 256 + threadIdx.x;             // N*128 threads
    if (t < NN * DD) agg[t] = 0.f;
    if (t < NN * HH) { denom[t] = 0.f; emaxU[t] = ENC_NEG_INF; }
}

// ---------------- kernel 4: e = att . leaky_relu(xl[src]+xr[dst]); atomic max per (dst,h) ----------------
__global__ void edge_logits_kernel(const float* __restrict__ xl, const float* __restrict__ xr,
                                   const long long* __restrict__ eidx,
                                   const float* __restrict__ att,
                                   float* __restrict__ eBuf, unsigned* __restrict__ emaxU) {
    long long tid = (long long)blockIdx.x * blockDim.x + threadIdx.x;
    if (tid >= (long long)EE * HH) return;
    int e = (int)(tid >> 3), h = (int)(tid & 7);
    int s, d; edge_sd(eidx, e, s, d);
    const float* pl = xl + (size_t)s * DD + h * CC;
    const float* pr = xr + (size_t)d * DD + h * CC;
    const float* pa = att + h * CC;
    float acc = 0.f;
#pragma unroll
    for (int i = 0; i < 16; i += 4) {
        float4 a = *(const float4*)(pl + i);
        float4 b = *(const float4*)(pr + i);
        float4 w = *(const float4*)(pa + i);
        acc += lrelu(a.x + b.x) * w.x + lrelu(a.y + b.y) * w.y +
               lrelu(a.z + b.z) * w.z + lrelu(a.w + b.w) * w.w;
    }
    eBuf[tid] = acc;
    atomicMax(emaxU + (size_t)d * HH + h, encF(acc));
}

// ---------------- kernel 5: ex = exp(e - emax[dst]); atomic sum per (dst,h) ----------------
__global__ void edge_softmax_kernel(const long long* __restrict__ eidx,
                                    float* __restrict__ eBuf,
                                    const unsigned* __restrict__ emaxU,
                                    float* __restrict__ denom) {
    long long tid = (long long)blockIdx.x * blockDim.x + threadIdx.x;
    if (tid >= (long long)EE * HH) return;
    int e = (int)(tid >> 3), h = (int)(tid & 7);
    int s, d; edge_sd(eidx, e, s, d); (void)s;
    float ex = __expf(eBuf[tid] - decF(emaxU[(size_t)d * HH + h]));
    eBuf[tid] = ex;
    atomicAdd(denom + (size_t)d * HH + h, ex);
}

// ---------------- kernel 6: agg[dst] += xl[src] * alpha ----------------
__global__ void edge_aggregate_kernel(const long long* __restrict__ eidx,
                                      const float* __restrict__ eBuf,
                                      const float* __restrict__ denom,
                                      const float* __restrict__ xl,
                                      float* __restrict__ agg) {
    long long tid = (long long)blockIdx.x * blockDim.x + threadIdx.x;
    if (tid >= (long long)EE * 32) return;              // (edge, head, quad-of-4-channels)
    int e = (int)(tid >> 5);
    int r = (int)(tid & 31);
    int h = r >> 2, q = r & 3;
    int s, d; edge_sd(eidx, e, s, d);
    float alpha = eBuf[(size_t)e * HH + h] / denom[(size_t)d * HH + h];
    const float4 v = *(const float4*)(xl + (size_t)s * DD + h * CC + q * 4);
    float* ag = agg + (size_t)d * DD + h * CC + q * 4;
    atomicAdd(ag + 0, v.x * alpha);
    atomicAdd(ag + 1, v.y * alpha);
    atomicAdd(ag + 2, v.z * alpha);
    atomicAdd(ag + 3, v.w * alpha);
}

// ---------------- kernel 7: bias + ELU + residual + LayerNorm (one wave32 per node) ----------------
__global__ void finalize_kernel(const float* __restrict__ agg, const float* __restrict__ bias,
                                const float* __restrict__ resid, const float* __restrict__ g,
                                const float* __restrict__ beta, float* __restrict__ out) {
    int node = blockIdx.x * (blockDim.x >> 5) + (threadIdx.x >> 5);
    if (node >= NN) return;
    int lane = threadIdx.x & 31;
    int c0 = lane * 4;
    const float* ag = agg + (size_t)node * DD + c0;
    const float* rs = resid + (size_t)node * DD + c0;
    float t[4];
    float s = 0.f, ss = 0.f;
#pragma unroll
    for (int j = 0; j < 4; ++j) {
        float v = elu1(ag[j] + bias[c0 + j]) + rs[j];
        t[j] = v; s += v; ss += v * v;
    }
#pragma unroll
    for (int off = 16; off > 0; off >>= 1) {
        s  += __shfl_xor(s,  off, 32);
        ss += __shfl_xor(ss, off, 32);
    }
    float mu  = s * (1.f / 128.f);
    float var = ss * (1.f / 128.f) - mu * mu;
    float inv = rsqrtf(var + LN_EPS);
    float* o = out + (size_t)node * DD + c0;
#pragma unroll
    for (int j = 0; j < 4; ++j)
        o[j] = (t[j] - mu) * inv * g[c0 + j] + beta[c0 + j];
}

// ---------------- host launcher ----------------
extern "C" void kernel_launch(void* const* d_in, const int* in_sizes, int n_in,
                              void* d_out, int out_size, void* d_ws, size_t ws_size,
                              hipStream_t stream) {
    (void)in_sizes; (void)n_in; (void)out_size; (void)ws_size;
    const float*     x    = (const float*)d_in[0];
    const long long* eidx = (const long long*)d_in[1];
    const float* Wl0 = (const float*)d_in[2],  *Wr0 = (const float*)d_in[3];
    const float* att0= (const float*)d_in[4],  *b0  = (const float*)d_in[5];
    const float* g0  = (const float*)d_in[6],  *be0 = (const float*)d_in[7];
    const float* Wl1 = (const float*)d_in[8],  *Wr1 = (const float*)d_in[9];
    const float* att1= (const float*)d_in[10], *b1  = (const float*)d_in[11];
    const float* g1  = (const float*)d_in[12], *be1 = (const float*)d_in[13];

    // carve workspace (512B-aligned regions)
    char* w = (char*)d_ws;
    auto carve = [&](size_t bytes) -> void* {
        void* p = (void*)w;
        w += (bytes + 511) & ~((size_t)511);
        return p;
    };
    float*          xl    = (float*)carve((size_t)NN * DD * 4);
    float*          xr    = (float*)carve((size_t)NN * DD * 4);
    float*          agg   = (float*)carve((size_t)NN * DD * 4);
    float*          x1    = (float*)carve((size_t)NN * DD * 4);
    float*          eBuf  = (float*)carve((size_t)EE * HH * 4);
    unsigned*       emaxU = (unsigned*)carve((size_t)NN * HH * 4);
    float*          denom = (float*)carve((size_t)NN * HH * 4);
    unsigned short* wpack = (unsigned short*)carve((size_t)4 * 16384 * 2);

    // pack all 4 weight matrices once
    pack_w_kernel<<<256, 256, 0, stream>>>(Wl0, Wr0, Wl1, Wr1, wpack);

    const int gemmBlocks = (NTILES + 7) / 8;             // 8 waves/block
    const long long eh   = (long long)EE * HH;
    const int ehBlocks   = (int)((eh + 255) / 256);      // 53125
    const int aggBlocks  = (int)(((long long)EE * 32 + 255) / 256); // 212500
    const int initBlocks = (NN * DD + 255) / 256;        // 50000
    const int finBlocks  = (NN + 7) / 8;                 // 12500

    for (int l = 0; l < 2; ++l) {
        const float* xin  = l ? x1   : x;
        const float* attL = l ? att1 : att0;
        const float* bL   = l ? b1   : b0;
        const float* gL   = l ? g1   : g0;
        const float* beL  = l ? be1  : be0;
        float*       outL = l ? (float*)d_out : x1;
        const unsigned short* wl = wpack + (size_t)(l * 2 + 0) * 16384;
        const unsigned short* wr = wpack + (size_t)(l * 2 + 1) * 16384;

        gemm_wmma_kernel<<<gemmBlocks, 256, 0, stream>>>(xin, wl, xl);
        gemm_wmma_kernel<<<gemmBlocks, 256, 0, stream>>>(xin, wr, xr);
        init_kernel<<<initBlocks, 256, 0, stream>>>(agg, denom, emaxU);
        edge_logits_kernel<<<ehBlocks, 256, 0, stream>>>(xl, xr, eidx, attL, eBuf, emaxU);
        edge_softmax_kernel<<<ehBlocks, 256, 0, stream>>>(eidx, eBuf, emaxU, denom);
        edge_aggregate_kernel<<<aggBlocks, 256, 0, stream>>>(eidx, eBuf, denom, xl, agg);
        finalize_kernel<<<finBlocks, 256, 0, stream>>>(agg, bL, xin, gL, beL, outL);
    }
}
